// Attention_62921270886427
// MI455X (gfx1250) — matrix operand order repaired and checked by reading, not compiled, not run on hardware
//
#include <hip/hip_runtime.h>
#include <hip/hip_bf16.h>
#include <math.h>

// Problem dims (fixed by the reference)
#define N_B   64
#define L_DIM 1024
#define H_DIM 1024
#define TL    64      // L-rows per workgroup in the score kernel
#define LDSTR 1032    // LDS A-row stride in bf16 elems (1024 + 8 pad -> conflict-free b128)

typedef __attribute__((ext_vector_type(16))) __bf16 v16bf;
typedef __attribute__((ext_vector_type(8)))  float  v8f;
typedef __attribute__((ext_vector_type(4)))  unsigned int v4u;

union Frag {            // trivially-constructible union (ext vectors only)
  v16bf v;
  v4u   q[2];
};

__device__ __forceinline__ unsigned short f2bf(float f) {
  unsigned u = __float_as_uint(f);
  unsigned r = 0x7FFFu + ((u >> 16) & 1u);   // round-to-nearest-even
  return (unsigned short)((u + r) >> 16);
}

// Branchless tanh: 1 - 2/(exp(2x)+1). Exact at +/-inf, no exec divergence;
// lowers to v_exp_f32 + fast-rcp + a few VALU ops (vs branchy ocml tanhf).
__device__ __forceinline__ float tanh_fast(float x) {
  float e = __expf(2.0f * x);
  return 1.0f - __fdividef(2.0f, e + 1.0f);
}

// ---------------------------------------------------------------------------
// Pack W_s (fp32 [j][k]) into bf16, pre-swizzled into WMMA B-fragment order:
// element index = (((jt*32 + ks)*32) + lane)*16 + e, where
//   j = jt*16 + (lane&15),  k = ks*32 + (lane&16 ? 16 : 0) + e
// Hot-loop B loads then become fully coalesced global_load_b128 pairs.
// ---------------------------------------------------------------------------
__global__ __launch_bounds__(256) void pack_ws_kernel(const float* __restrict__ Ws,
                                                      unsigned short* __restrict__ Wp) {
  int idx  = blockIdx.x * 256 + threadIdx.x;   // 0 .. 1M-1
  int e    = idx & 15;
  int lane = (idx >> 4) & 31;
  int ks   = (idx >> 9) & 31;
  int jt   = idx >> 14;
  int j = jt * 16 + (lane & 15);
  int k = ks * 32 + ((lane & 16) ? 16 : 0) + e;
  Wp[idx] = f2bf(Ws[(size_t)j * H_DIM + k]);
}

// ---------------------------------------------------------------------------
// dh[n][j] = sum_k dec[n][k] * W_h[j][k]   (fp32, tiny: 0.13 GFLOP)
// One block per n; each wave owns a j, lanes stride over k (coalesced W_h).
// ---------------------------------------------------------------------------
__global__ __launch_bounds__(256) void dh_kernel(const float* __restrict__ dec,
                                                 const float* __restrict__ Wh,
                                                 float* __restrict__ dh) {
  __shared__ float ds[H_DIM];
  int n = blockIdx.x, tid = threadIdx.x;
  for (int i = tid; i < H_DIM; i += 256) ds[i] = dec[(size_t)n * H_DIM + i];
  __syncthreads();
  int wave = tid >> 5, lane = tid & 31;
  for (int j = wave; j < H_DIM; j += 8) {
    const float* wr = Wh + (size_t)j * H_DIM;
    float p = 0.f;
    for (int k = lane; k < H_DIM; k += 32) p += wr[k] * ds[k];
#pragma unroll
    for (int o = 16; o >= 1; o >>= 1) p += __shfl_xor(p, o, 32);
    if (lane == 0) dh[(size_t)n * H_DIM + j] = p;
  }
}

// ---------------------------------------------------------------------------
// Fused score kernel: per WG (one n, 64 L-rows)
//   eh_tile = E_tile @ Ws^T  (bf16 WMMA, f32 acc)  -> tanh(+dh) * v -> row sums
// Never materializes eh in HBM. 8 waves x 8 col-tiles; B fragment reused
// across 4 row-tiles per k-step. Per-lane score accumulation across all
// column tiles; a single 16-lane xor reduction at the very end. Deterministic
// fixed-order LDS partial-sum reduction (no float atomics).
// ---------------------------------------------------------------------------
__global__ __launch_bounds__(256) void score_kernel(
    const float* __restrict__ E, const unsigned short* __restrict__ Wp,
    const float* __restrict__ dhg, const float* __restrict__ vg,
    float* __restrict__ scores) {
  __shared__ unsigned short As[TL * LDSTR];  // 132 KB bf16 A tile (padded rows)
  __shared__ float dh_s[H_DIM];
  __shared__ float v_s[H_DIM];
  __shared__ float part_s[8 * TL];           // per-wave score partials

  int tid = threadIdx.x;
  int n  = blockIdx.x >> 4;                  // 16 L-blocks per n
  int l0 = (blockIdx.x & 15) * TL;

  // Stage E tile: fp32 global -> bf16 LDS (pairs packed into u32 stores)
  const float* Erow = E + ((size_t)n * L_DIM + l0) * H_DIM;
  unsigned* As32 = (unsigned*)As;
  for (int i = tid; i < TL * 512; i += 256) {
    int row = i >> 9, pc = i & 511;
    float2 xy = *(const float2*)(Erow + (size_t)row * H_DIM + pc * 2);
    As32[row * (LDSTR / 2) + pc] = (unsigned)f2bf(xy.x) | ((unsigned)f2bf(xy.y) << 16);
  }
  for (int i = tid; i < H_DIM; i += 256) {
    dh_s[i] = dhg[(size_t)n * H_DIM + i];
    v_s[i]  = vg[i];
  }
  __syncthreads();

  int wave = tid >> 5, lane = tid & 31;
  int mlo = lane & 15;
  int hi  = (lane >> 4) & 1;

  // Per-lane partial scores: element (m = rt*16 + 8*hi + r, this lane's j's)
  float sAcc[4][8];
#pragma unroll
  for (int rt = 0; rt < 4; ++rt)
#pragma unroll
    for (int r = 0; r < 8; ++r) sAcc[rt][r] = 0.f;

  const v8f zero = {0.f,0.f,0.f,0.f,0.f,0.f,0.f,0.f};

  for (int jt = wave; jt < 64; jt += 8) {    // 8 column tiles per wave
    v8f acc[4];
#pragma unroll
    for (int rt = 0; rt < 4; ++rt) acc[rt] = zero;

    for (int ks = 0; ks < 32; ++ks) {        // K loop: 1024 / 32
      // B fragment: pre-packed, coalesced (two b128 per lane)
      Frag bf_;
      const v4u* bq = (const v4u*)Wp + ((size_t)(jt * 32 + ks) * 32 + lane) * 2;
      bf_.q[0] = bq[0];
      bf_.q[1] = bq[1];
      int kb = ks * 32 + (hi ? 16 : 0);      // lanes 16-31 start at K+16 (contiguous, matches packed B)
#pragma unroll
      for (int rt = 0; rt < 4; ++rt) {       // reuse B across 4 row tiles
        Frag af;
        const unsigned short* ar = As + (rt * 16 + mlo) * LDSTR + kb;
        af.q[0] = *(const v4u*)(ar);         // K = kb .. kb+7
        af.q[1] = *(const v4u*)(ar + 8);     // K = kb+8 .. kb+15
        acc[rt] = __builtin_amdgcn_wmma_f32_16x16x32_bf16(
            false, af.v, false, bf_.v, (short)0, acc[rt], false, false);
      }
    }

    // Epilogue: tanh(eh + dh) * v, accumulated per-lane (no shuffles here)
    int j = jt * 16 + mlo;
    float dhv = dh_s[j];
    float vv  = v_s[j];
#pragma unroll
    for (int rt = 0; rt < 4; ++rt) {
#pragma unroll
      for (int r = 0; r < 8; ++r)
        sAcc[rt][r] += tanh_fast(acc[rt][r] + dhv) * vv;
    }
  }

  // Single cross-lane reduction over the 16 columns per half-wave
#pragma unroll
  for (int rt = 0; rt < 4; ++rt) {
#pragma unroll
    for (int r = 0; r < 8; ++r) {
      float t = sAcc[rt][r];
      t += __shfl_xor(t, 1, 32);
      t += __shfl_xor(t, 2, 32);
      t += __shfl_xor(t, 4, 32);
      t += __shfl_xor(t, 8, 32);
      sAcc[rt][r] = t;
    }
  }
  if (mlo == 0) {
#pragma unroll
    for (int rt = 0; rt < 4; ++rt)
#pragma unroll
      for (int r = 0; r < 8; ++r)
        part_s[wave * TL + rt * 16 + hi * 8 + r] = sAcc[rt][r];
  }
  __syncthreads();
  if (tid < TL) {                            // deterministic fixed-order sum
    float s = 0.f;
#pragma unroll
    for (int w = 0; w < 8; ++w) s += part_s[w * TL + tid];
    scores[(size_t)n * L_DIM + l0 + tid] = s;
  }
}

// ---------------------------------------------------------------------------
// Masked softmax over L per batch row. Block per n, 4 elements/thread.
// ---------------------------------------------------------------------------
__global__ __launch_bounds__(256) void softmax_kernel(const float* __restrict__ scores,
                                                      const int* __restrict__ mask,
                                                      float* __restrict__ attn) {
  __shared__ float redm[8];
  __shared__ float reds[8];
  __shared__ float bc[2];
  int n = blockIdx.x, tid = threadIdx.x;
  int wave = tid >> 5, lane = tid & 31;
  float s[4];
#pragma unroll
  for (int c = 0; c < 4; ++c) {
    int l = c * 256 + tid;
    float sc = scores[(size_t)n * L_DIM + l];
    s[c] = mask[(size_t)n * L_DIM + l] ? -INFINITY : sc;
  }
  float m = fmaxf(fmaxf(s[0], s[1]), fmaxf(s[2], s[3]));
#pragma unroll
  for (int o = 16; o >= 1; o >>= 1) m = fmaxf(m, __shfl_xor(m, o, 32));
  if (lane == 0) redm[wave] = m;
  __syncthreads();
  if (tid == 0) {
    float mm = redm[0];
#pragma unroll
    for (int w = 1; w < 8; ++w) mm = fmaxf(mm, redm[w]);
    bc[0] = mm;
  }
  __syncthreads();
  float gmax = bc[0];
  float e[4], psum = 0.f;
#pragma unroll
  for (int c = 0; c < 4; ++c) { e[c] = __expf(s[c] - gmax); psum += e[c]; }
#pragma unroll
  for (int o = 16; o >= 1; o >>= 1) psum += __shfl_xor(psum, o, 32);
  if (lane == 0) reds[wave] = psum;
  __syncthreads();
  if (tid == 0) {
    float ss = 0.f;
#pragma unroll
    for (int w = 0; w < 8; ++w) ss += reds[w];
    bc[1] = ss;
  }
  __syncthreads();
  float inv = 1.0f / bc[1];
#pragma unroll
  for (int c = 0; c < 4; ++c)
    attn[(size_t)n * L_DIM + c * 256 + tid] = e[c] * inv;
}

// ---------------------------------------------------------------------------
// context[n][h] = sum_l attn[n][l] * E[n][l][h]. Pure bandwidth: one block
// per n, float4 streams of E (coalesced b128), attn staged in LDS.
// ---------------------------------------------------------------------------
__global__ __launch_bounds__(256) void context_kernel(const float* __restrict__ E,
                                                      const float* __restrict__ attn,
                                                      float* __restrict__ ctx) {
  __shared__ float a_s[L_DIM];
  int n = blockIdx.x, tid = threadIdx.x;
  for (int i = tid; i < L_DIM; i += 256) a_s[i] = attn[(size_t)n * L_DIM + i];
  __syncthreads();
  const float4* Ep = (const float4*)(E + (size_t)n * L_DIM * H_DIM);
  float ax = 0.f, ay = 0.f, az = 0.f, aw = 0.f;
  for (int l = 0; l < L_DIM; ++l) {
    float al = a_s[l];
    float4 ev = Ep[(size_t)l * (H_DIM / 4) + tid];
    ax += al * ev.x; ay += al * ev.y; az += al * ev.z; aw += al * ev.w;
  }
  float4* cp = (float4*)(ctx + (size_t)n * H_DIM);
  cp[tid] = make_float4(ax, ay, az, aw);
}

// ---------------------------------------------------------------------------
extern "C" void kernel_launch(void* const* d_in, const int* in_sizes, int n_in,
                              void* d_out, int out_size, void* d_ws, size_t ws_size,
                              hipStream_t stream) {
  const float* dec  = (const float*)d_in[0];   // [64,1024]
  const float* E    = (const float*)d_in[1];   // [64,1024,1024]
  const int*   mask = (const int*)d_in[2];     // [64,1024]
  const float* Wh   = (const float*)d_in[3];   // [1024,1024]
  const float* Ws   = (const float*)d_in[4];   // [1024,1024]
  const float* v    = (const float*)d_in[5];   // [1024]

  float* ctx  = (float*)d_out;                 // output 0: context [64,1024]
  float* attn = (float*)d_out + N_B * H_DIM;   // output 1: attn    [64,1024]

  // Workspace layout: 2 MB packed bf16 W_s | 256 KB dh | 256 KB scores
  unsigned short* Wp  = (unsigned short*)d_ws;
  float* dhW  = (float*)((char*)d_ws + 2u * 1024u * 1024u);
  float* scrW = dhW + N_B * H_DIM;

  pack_ws_kernel<<<(H_DIM * H_DIM) / 256, 256, 0, stream>>>(Ws, Wp);
  dh_kernel<<<N_B, 256, 0, stream>>>(dec, Wh, dhW);
  score_kernel<<<N_B * (L_DIM / TL), 256, 0, stream>>>(E, Wp, dhW, v, scrW);
  softmax_kernel<<<N_B, 256, 0, stream>>>(scrW, mask, attn);
  context_kernel<<<N_B, 256, 0, stream>>>(E, attn, ctx);
}
